// EAA_54743653155217
// MI455X (gfx1250) — compile-verified, matrix operand order checked by reading, not run on hardware
//
#include <hip/hip_runtime.h>

typedef __attribute__((ext_vector_type(16))) __bf16 v16bf;
typedef __attribute__((ext_vector_type(8)))  __bf16 v8bf;
typedef __attribute__((ext_vector_type(8)))  float  v8f;
typedef __attribute__((ext_vector_type(4)))  int    v4i;

#define EPSN 1e-12f
#define XSTR 520  // padded LDS row stride (elements) -> rows land on distinct banks

#if __has_builtin(__builtin_amdgcn_global_load_async_to_lds_b128) && \
    __has_builtin(__builtin_amdgcn_s_wait_asynccnt)
#define HAS_ASYNC_LDS 1
#else
#define HAS_ASYNC_LDS 0
#endif

#if HAS_ASYNC_LDS
typedef __attribute__((address_space(1))) v4i* gasp;  // global b128 pointer
typedef __attribute__((address_space(3))) v4i* lasp;  // LDS b128 pointer
__device__ __forceinline__ void async_cp16(const void* g, void* l) {
  __builtin_amdgcn_global_load_async_to_lds_b128((gasp)g, (lasp)l, 0, 0);
}
__device__ __forceinline__ void async_wait0() {
  __builtin_amdgcn_s_wait_asynccnt(0);
}
#endif

__device__ __forceinline__ v16bf cat8(v8bf lo, v8bf hi) {
  return __builtin_shufflevector(lo, hi, 0,1,2,3,4,5,6,7,8,9,10,11,12,13,14,15);
}

// ---------------- weight transpose + bf16 convert: dst[n*K + k] = src[k*N + n]
__global__ __launch_bounds__(256) void EAA_cvtT(const float* __restrict__ src,
                                                __bf16* __restrict__ dst,
                                                int K, int N) {
  int i = blockIdx.x * 256 + threadIdx.x;
  if (i < K * N) {
    int k = i / N, n = i - k * N;
    dst[n * K + k] = (__bf16)src[i];
  }
}

__global__ __launch_bounds__(256) void EAA_zero(float* __restrict__ P,
                                                float* __restrict__ S) {
  int i = blockIdx.x * 256 + threadIdx.x;
  if (i < 4096) P[i] = 0.f;
  if (i < 8)    S[i] = 0.f;
}

// ---------------- kernel 1: Q/K projection + L2 norm + pooled-context partials
__global__ __launch_bounds__(256) void EAA_qk(
    const float* __restrict__ x,
    const __bf16* __restrict__ qwt, const __bf16* __restrict__ kwt,
    const float* __restrict__ qb,  const float* __restrict__ kb,
    const float* __restrict__ wg,
    __bf16* __restrict__ Qbf, __bf16* __restrict__ Kbf,
    float* __restrict__ P, float* __restrict__ S)
{
  __shared__ __bf16 xs[32 * XSTR];
  __shared__ float rnq[32], rnk[32], rdot[32], a_sh[32];

  const int tid  = threadIdx.x;
  const int lane = tid & 31;
  const int wave = tid >> 5;
  const int tok0 = blockIdx.x * 32;
  const int b    = tok0 >> 12;   // 4096 tokens per batch

#if HAS_ASYNC_LDS
  // stage x tile via async DMA (global f32 -> LDS f32), then convert to bf16
  __shared__ float xf[32 * 512];
  for (int i = tid * 4; i < 32 * 512; i += 256 * 4)
    async_cp16(x + (size_t)tok0 * 512 + i, &xf[i]);
  async_wait0();   // per-wave: each thread converts only its own chunks
  for (int i = tid * 4; i < 32 * 512; i += 256 * 4) {
    int tr = i >> 9, d = i & 511;
    float4 v = *(const float4*)(&xf[i]);
    __bf16* p = &xs[tr * XSTR + d];
    p[0] = (__bf16)v.x; p[1] = (__bf16)v.y; p[2] = (__bf16)v.z; p[3] = (__bf16)v.w;
  }
#else
  for (int i = tid * 4; i < 32 * 512; i += 256 * 4) {
    int tr = i >> 9, d = i & 511;
    float4 v = *(const float4*)(x + (size_t)(tok0 + tr) * 512 + d);
    __bf16* p = &xs[tr * XSTR + d];
    p[0] = (__bf16)v.x; p[1] = (__bf16)v.y; p[2] = (__bf16)v.z; p[3] = (__bf16)v.w;
  }
#endif
  if (tid < 32) { rnq[tid] = 0.f; rnk[tid] = 0.f; rdot[tid] = 0.f; }
  __syncthreads();

  const int mh    = wave & 1;          // row half: 0 -> rows 0..15, 1 -> 16..31
  const int nq    = wave >> 1;         // col slab * 128
  const int row   = mh * 16 + (lane & 15);
  const int n0    = nq * 128;
  const int khalf = (lane >> 4) * 8;   // fragment K sub-offset per lane half

  v8f qacc[8] = {}; v8f kacc[8] = {};

  for (int kk = 0; kk < 16; ++kk) {
    const int kbase = kk * 32 + khalf;
    v16bf a = cat8(*(const v8bf*)(&xs[row * XSTR + kbase]),
                   *(const v8bf*)(&xs[row * XSTR + kbase + 16]));
    #pragma unroll
    for (int t = 0; t < 8; ++t) {
      const int n = n0 + t * 16 + (lane & 15);
      const __bf16* bq = qwt + (size_t)n * 512 + kbase;
      v16bf bfq = cat8(*(const v8bf*)bq, *(const v8bf*)(bq + 16));
      qacc[t] = __builtin_amdgcn_wmma_f32_16x16x32_bf16(false, a, false, bfq,
                                                        (short)0, qacc[t], false, false);
      const __bf16* bk = kwt + (size_t)n * 512 + kbase;
      v16bf bfk = cat8(*(const v8bf*)bk, *(const v8bf*)(bk + 16));
      kacc[t] = __builtin_amdgcn_wmma_f32_16x16x32_bf16(false, a, false, bfk,
                                                        (short)0, kacc[t], false, false);
    }
  }

  // bias
  #pragma unroll
  for (int t = 0; t < 8; ++t) {
    const int n = n0 + t * 16 + (lane & 15);
    float qbn = qb[n], kbn = kb[n];
    #pragma unroll
    for (int r = 0; r < 8; ++r) { qacc[t][r] += qbn; kacc[t][r] += kbn; }
  }
  // row-wise squared sums (per wave partial over its 128 cols)
  #pragma unroll
  for (int r = 0; r < 8; ++r) {
    const int m = mh * 16 + r + 8 * (lane >> 4);
    float sq = 0.f, sk = 0.f;
    #pragma unroll
    for (int t = 0; t < 8; ++t) {
      float q = qacc[t][r]; sq += q * q;
      float k2 = kacc[t][r]; sk += k2 * k2;
    }
    atomicAdd(&rnq[m], sq);
    atomicAdd(&rnk[m], sk);
  }
  __syncthreads();

  // normalize, store bf16 Q/K, accumulate row-dot with w_g
  float wgn[8];
  #pragma unroll
  for (int t = 0; t < 8; ++t) wgn[t] = wg[n0 + t * 16 + (lane & 15)];
  #pragma unroll
  for (int r = 0; r < 8; ++r) {
    const int m = mh * 16 + r + 8 * (lane >> 4);
    const size_t tokm = (size_t)(tok0 + m);
    float iq = 1.f / fmaxf(sqrtf(rnq[m]), EPSN);
    float ik = 1.f / fmaxf(sqrtf(rnk[m]), EPSN);
    float rd = 0.f;
    #pragma unroll
    for (int t = 0; t < 8; ++t) {
      const int n = n0 + t * 16 + (lane & 15);
      float q  = qacc[t][r] * iq; qacc[t][r] = q;
      float k2 = kacc[t][r] * ik;
      Qbf[tokm * 512 + n] = (__bf16)q;
      Kbf[tokm * 512 + n] = (__bf16)k2;
      rd += q * wgn[t];
    }
    atomicAdd(&rdot[m], rd);
  }
  __syncthreads();

  if (tid < 32) {
    float a = rdot[tid] * 0.0625f;   // * 256^-0.5
    a_sh[tid] = a;
    atomicAdd(&S[b], a * a);
  }
  __syncthreads();

  // P[b][n] += sum_m a_m * Q[m][n]
  #pragma unroll
  for (int t = 0; t < 8; ++t) {
    const int n = n0 + t * 16 + (lane & 15);
    float acc = 0.f;
    #pragma unroll
    for (int r = 0; r < 8; ++r) {
      const int m = mh * 16 + r + 8 * (lane >> 4);
      acc += a_sh[m] * qacc[t][r];
    }
    atomicAdd(&P[b * 512 + n], acc);
  }
}

// ---------------- kernel 2: G = P / max(sqrt(S), eps)
__global__ __launch_bounds__(256) void EAA_gctx(const float* __restrict__ P,
                                                const float* __restrict__ S,
                                                float* __restrict__ G) {
  int i = blockIdx.x * 256 + threadIdx.x;
  if (i < 8 * 512) {
    int b = i >> 9;
    G[i] = P[i] / fmaxf(sqrtf(S[b]), EPSN);
  }
}

// ---------------- kernel 3: out = ((G*K)@proj + pb + Q) @ outw + ob
__global__ __launch_bounds__(256) void EAA_out(
    const __bf16* __restrict__ Qbf, const __bf16* __restrict__ Kbf,
    const float* __restrict__ G,
    const __bf16* __restrict__ pwt, const float* __restrict__ pb,
    const __bf16* __restrict__ owt, const float* __restrict__ ob,
    float* __restrict__ out)
{
  __shared__ __bf16 gs[32 * XSTR];   // gated = G (.) K
  __shared__ __bf16 qs[32 * XSTR];   // Q residual tile
  __shared__ __bf16 rs[32 * XSTR];   // K staging (async), then res (bf16)

  const int tid  = threadIdx.x;
  const int lane = tid & 31;
  const int wave = tid >> 5;
  const int tok0 = blockIdx.x * 32;
  const int b    = tok0 >> 12;

#if HAS_ASYNC_LDS
  // DMA the K tile (into rs, reused later) and the Q tile straight to LDS
  for (int i = tid * 8; i < 32 * 512; i += 256 * 8) {
    int tr = i >> 9, d = i & 511;
    async_cp16(Kbf + (size_t)(tok0 + tr) * 512 + d, &rs[tr * XSTR + d]);
    async_cp16(Qbf + (size_t)(tok0 + tr) * 512 + d, &qs[tr * XSTR + d]);
  }
  async_wait0();   // per-wave: gating loop below touches only this wave's chunks
  for (int i = tid * 8; i < 32 * 512; i += 256 * 8) {
    int tr = i >> 9, d = i & 511;
    v8bf kv = *(const v8bf*)(&rs[tr * XSTR + d]);
    v8bf gv;
    #pragma unroll
    for (int j = 0; j < 8; ++j)
      gv[j] = (__bf16)((float)kv[j] * G[b * 512 + d + j]);
    *(v8bf*)(&gs[tr * XSTR + d]) = gv;
  }
#else
  for (int i = tid * 8; i < 32 * 512; i += 256 * 8) {
    int tr = i >> 9, d = i & 511;
    v8bf kv = *(const v8bf*)(Kbf + (size_t)(tok0 + tr) * 512 + d);
    v8bf qv = *(const v8bf*)(Qbf + (size_t)(tok0 + tr) * 512 + d);
    v8bf gv;
    #pragma unroll
    for (int j = 0; j < 8; ++j)
      gv[j] = (__bf16)((float)kv[j] * G[b * 512 + d + j]);
    *(v8bf*)(&gs[tr * XSTR + d]) = gv;
    *(v8bf*)(&qs[tr * XSTR + d]) = qv;
  }
#endif
  __syncthreads();

  const int mh    = wave & 1;
  const int nq    = wave >> 1;
  const int row   = mh * 16 + (lane & 15);
  const int khalf = (lane >> 4) * 8;

  { // GEMM1: res = gated @ proj_w  (+pb +Q), slab 16x128 per wave
    const int n0 = nq * 128;
    v8f acc[8] = {};
    for (int kk = 0; kk < 16; ++kk) {
      const int kb2 = kk * 32 + khalf;
      v16bf a = cat8(*(const v8bf*)(&gs[row * XSTR + kb2]),
                     *(const v8bf*)(&gs[row * XSTR + kb2 + 16]));
      #pragma unroll
      for (int t = 0; t < 8; ++t) {
        const int n = n0 + t * 16 + (lane & 15);
        const __bf16* bp = pwt + (size_t)n * 512 + kb2;
        v16bf bb = cat8(*(const v8bf*)bp, *(const v8bf*)(bp + 16));
        acc[t] = __builtin_amdgcn_wmma_f32_16x16x32_bf16(false, a, false, bb,
                                                         (short)0, acc[t], false, false);
      }
    }
    __syncthreads();   // rs (K staging) is dead; safe to overwrite with res
    #pragma unroll
    for (int t = 0; t < 8; ++t) {
      const int n = n0 + t * 16 + (lane & 15);
      float pbn = pb[n];
      #pragma unroll
      for (int r = 0; r < 8; ++r) {
        const int m = mh * 16 + r + 8 * (lane >> 4);
        float v = acc[t][r] + pbn + (float)qs[m * XSTR + n];
        rs[m * XSTR + n] = (__bf16)v;
      }
    }
  }
  __syncthreads();

  { // GEMM2: out = res @ out_w (+ob), slab 16x64 per wave
    const int n0 = nq * 64;
    v8f acc[4] = {};
    for (int kk = 0; kk < 16; ++kk) {
      const int kb2 = kk * 32 + khalf;
      v16bf a = cat8(*(const v8bf*)(&rs[row * XSTR + kb2]),
                     *(const v8bf*)(&rs[row * XSTR + kb2 + 16]));
      #pragma unroll
      for (int t = 0; t < 4; ++t) {
        const int n = n0 + t * 16 + (lane & 15);
        const __bf16* bo = owt + (size_t)n * 512 + kb2;
        v16bf bb = cat8(*(const v8bf*)bo, *(const v8bf*)(bo + 16));
        acc[t] = __builtin_amdgcn_wmma_f32_16x16x32_bf16(false, a, false, bb,
                                                         (short)0, acc[t], false, false);
      }
    }
    #pragma unroll
    for (int t = 0; t < 4; ++t) {
      const int n = n0 + t * 16 + (lane & 15);
      float obn = ob[n];
      #pragma unroll
      for (int r = 0; r < 8; ++r) {
        const int m = mh * 16 + r + 8 * (lane >> 4);
        out[(size_t)(tok0 + m) * 256 + n] = acc[t][r] + obn;
      }
    }
  }
}

extern "C" void kernel_launch(void* const* d_in, const int* in_sizes, int n_in,
                              void* d_out, int out_size, void* d_ws, size_t ws_size,
                              hipStream_t stream) {
  const float* x      = (const float*)d_in[0];
  const float* q_w    = (const float*)d_in[1];
  const float* q_b    = (const float*)d_in[2];
  const float* k_w    = (const float*)d_in[3];
  const float* k_b    = (const float*)d_in[4];
  const float* w_g    = (const float*)d_in[5];
  const float* proj_w = (const float*)d_in[6];
  const float* proj_b = (const float*)d_in[7];
  const float* out_w  = (const float*)d_in[8];
  const float* out_b  = (const float*)d_in[9];

  char* ws = (char*)d_ws;
  __bf16* qwt = (__bf16*)(ws);                       // 512 KB
  __bf16* kwt = (__bf16*)(ws + 524288);              // 512 KB
  __bf16* pwt = (__bf16*)(ws + 1048576);             // 512 KB
  __bf16* owt = (__bf16*)(ws + 1572864);             // 256 KB
  float*  P   = (float*)(ws + 1835008);              // 16 KB
  float*  S   = (float*)(ws + 1851392);              // 32 B
  float*  G   = (float*)(ws + 1851648);              // 16 KB
  __bf16* Qbf = (__bf16*)(ws + (size_t)2097152);               // 32 MB
  __bf16* Kbf = (__bf16*)(ws + (size_t)2097152 + 33554432);    // 32 MB

  EAA_cvtT<<<(512 * 512 + 255) / 256, 256, 0, stream>>>(q_w, qwt, 512, 512);
  EAA_cvtT<<<(512 * 512 + 255) / 256, 256, 0, stream>>>(k_w, kwt, 512, 512);
  EAA_cvtT<<<(512 * 512 + 255) / 256, 256, 0, stream>>>(proj_w, pwt, 512, 512);
  EAA_cvtT<<<(512 * 256 + 255) / 256, 256, 0, stream>>>(out_w, owt, 512, 256);
  EAA_zero<<<16, 256, 0, stream>>>(P, S);

  EAA_qk<<<1024, 256, 0, stream>>>(x, qwt, kwt, q_b, k_b, w_g, Qbf, Kbf, P, S);
  EAA_gctx<<<16, 256, 0, stream>>>(P, S, G);
  EAA_out<<<1024, 256, 0, stream>>>(Qbf, Kbf, G, pwt, proj_b, owt, out_b,
                                    (float*)d_out);
}
